// NeuralODELSTM_78185584656870
// MI455X (gfx1250) — matrix operand order repaired and checked by reading, not compiled, not run on hardware
//
#include <hip/hip_runtime.h>
#include <hip/hip_bf16.h>
#include <cstddef>
#include <cstdint>

#define BATCH 512
#define SEQ   256
#define INF   64
#define HID   128
#define GDIM  512   // 4*HID

typedef __bf16 bf16_t;
typedef __attribute__((ext_vector_type(16))) __bf16 v16bf;
typedef __attribute__((ext_vector_type(8)))  __bf16 v8bf;
typedef __attribute__((ext_vector_type(4)))  __bf16 v4bf;
typedef __attribute__((ext_vector_type(8)))  float  v8f;

// ---------------------------------------------------------------------------
// Opaque zero index: empty asm redefines a 0 every call, so weight-fragment
// load addresses are not loop-invariant across the sequence loop (no LICM ->
// no scratch spills), while the base pointer keeps its global address space
// (loads stay global_load_b128 off an SGPR base, not flat_load).
// ---------------------------------------------------------------------------
__device__ __forceinline__ int opaque0() {
  int z = 0;
  asm volatile("" : "+s"(z));
  return z;
}

// ---------------------------------------------------------------------------
// Weight packing: W is (Nout x Kin) row-major f32; we emit bf16 fragments of
// B = W^T (Kpad x Nout) in exact WMMA B-operand layout (16x16x32 bf16):
//   fragment f = nt*numKt + kt, 512 bf16 elems, element (lane,e):
//     k = kt*32 + 8*(lane>=16) + (e&7) + 16*(e>=8),  n = nt*16 + (lane&15)
// Rows k >= Kin are zero (used to pad the [h,t] K=129 matmul to K=160).
// ---------------------------------------------------------------------------
__global__ void pack_wT(const float* __restrict__ W, bf16_t* __restrict__ dst,
                        int Nout, int Kin, int Kpad) {
  const int numKt = Kpad >> 5;
  const int total = (Nout >> 4) * numKt * 512;
  for (int idx = blockIdx.x * blockDim.x + threadIdx.x; idx < total;
       idx += gridDim.x * blockDim.x) {
    const int e    = idx & 15;
    const int lane = (idx >> 4) & 31;
    const int f    = idx >> 9;
    const int kt   = f % numKt;
    const int nt   = f / numKt;
    const int k = kt * 32 + ((lane >> 4) << 3) + (e & 7) + ((e >> 3) << 4);
    const int n = nt * 16 + (lane & 15);
    const float v = (k < Kin) ? W[n * Kin + k] : 0.0f;
    dst[idx] = (bf16_t)v;
  }
}

// ---------------------------------------------------------------------------
// Fragment helpers
// ---------------------------------------------------------------------------
// A operand (16x32 bf16, MxK) from row-major LDS tile with row stride ldm.
// ISA layout: lanes 0-15: row=lane, elems {K0..7, K16..23};
//             lanes 16-31: row=lane-16, elems {K8..15, K24..31}.
__device__ __forceinline__ v16bf load_a_frag(const bf16_t* tile, int ldm,
                                             int k0, int lane) {
  const int r    = lane & 15;
  const int koff = (lane >> 4) << 3;           // 0 or 8
  const bf16_t* p = tile + r * ldm + k0 + koff;
  v8bf lo = *(const v8bf*)(p);                 // K = k0+koff+0..7
  v8bf hi = *(const v8bf*)(p + 16);            // K = k0+koff+16..23
  return __builtin_shufflevector(lo, hi, 0, 1, 2, 3, 4, 5, 6, 7,
                                 8, 9, 10, 11, 12, 13, 14, 15);
}

// B operand from fragment-packed weights (global or LDS): one 32B load/lane.
// zoff is an extra element offset (opaque 0 for global weights).
__device__ __forceinline__ v16bf load_b_frag(const bf16_t* packed, int zoff,
                                             int numKt, int kt, int nt,
                                             int lane) {
  const bf16_t* p =
      packed + zoff + (((size_t)(nt * numKt + kt)) << 9) + (lane << 4);
  return *(const v16bf*)(p);
}

__device__ __forceinline__ float sigmoidf_(float v) {
  return 1.0f / (1.0f + __expf(-v));
}

// Store C/D fragment (+per-column bias) into row-major f32 LDS buffer.
__device__ __forceinline__ void store_acc(float* buf, int ldm, int n0, int lane,
                                          const v8f& acc, float bias) {
  const int col = n0 + (lane & 15);
  const int rb  = (lane >> 4) << 3;
#pragma unroll
  for (int r = 0; r < 8; ++r)
    buf[(rb + r) * ldm + col] = acc[r] + bias;
}

// ---------------------------------------------------------------------------
// LSTM gate GEMM: G(16x512) = Xin(16xKx)*WxT + H(16x128)*WhT + bih + bhh
// 32 column tiles, 4 per wave. Kx = 64 (layer0) or 128 (layer1).
// A fragments are loaded once into registers and reused for all 4 col tiles.
// ---------------------------------------------------------------------------
__device__ __forceinline__ void lstm_gates(const bf16_t* sXin, int ldx, int nkx,
                                           const bf16_t* __restrict__ WxT,
                                           const bf16_t* sHin,
                                           const bf16_t* __restrict__ WhT,
                                           const float* __restrict__ bih,
                                           const float* __restrict__ bhh,
                                           float* G, int wave, int lane) {
  const int zx = opaque0();
  const int zh = opaque0();

  v16bf ax[4];
#pragma unroll
  for (int kt = 0; kt < 4; ++kt)
    if (kt < nkx) ax[kt] = load_a_frag(sXin, ldx, kt * 32, lane);
  v16bf ah[4];
#pragma unroll
  for (int kt = 0; kt < 4; ++kt)
    ah[kt] = load_a_frag(sHin, HID, kt * 32, lane);

#pragma unroll
  for (int c = 0; c < 4; ++c) {
    const int nt = wave * 4 + c;
    v8f acc = {};
#pragma unroll
    for (int kt = 0; kt < 4; ++kt) {
      if (kt < nkx) {
        v16bf b = load_b_frag(WxT, zx, nkx, kt, nt, lane);
        acc = __builtin_amdgcn_wmma_f32_16x16x32_bf16(
            false, ax[kt], false, b, (short)0, acc, false, false);
      }
    }
#pragma unroll
    for (int kt = 0; kt < 4; ++kt) {
      v16bf b = load_b_frag(WhT, zh, 4, kt, nt, lane);
      acc = __builtin_amdgcn_wmma_f32_16x16x32_bf16(
          false, ah[kt], false, b, (short)0, acc, false, false);
    }
    const int col = nt * 16 + (lane & 15);
    store_acc(G, GDIM, nt * 16, lane, acc, bih[col] + bhh[col]);
  }
}

// Gate activations + cell update; writes h (f32), c (f32), and bf16 h into the
// fnode A-staging buffer (row stride 160, cols 0..127).
__device__ __forceinline__ void lstm_elem(const float* G, float* h, float* c,
                                          bf16_t* sA, int tid) {
  for (int i = tid; i < 16 * HID; i += 256) {
    const int m = i >> 7, j = i & 127;
    const float gi = G[m * GDIM + j];
    const float gf = G[m * GDIM + HID + j];
    const float gg = G[m * GDIM + 2 * HID + j];
    const float go = G[m * GDIM + 3 * HID + j];
    const float c2 = sigmoidf_(gf) * c[i] + sigmoidf_(gi) * tanhf(gg);
    const float h2 = sigmoidf_(go) * tanhf(c2);
    c[i] = c2;
    h[i] = h2;
    sA[m * 160 + j] = (bf16_t)h2;
  }
}

// ---------------------------------------------------------------------------
// One RK4 step of dh/dt = W2*tanh(W1*[h,t] + b1) + b2 over [t0, t1].
// sA (16x160 bf16) holds [h_work | t | 0-pad]; K=160 -> 5 WMMA k-tiles.
// Each of the 8 waves owns one 16-wide output column tile.
// W1T fragments (global) are invariant across the 4 stages -> hoisted into
// registers; W2T fragments come from LDS (staged once at kernel start).
// On exit: hbase (f32 LDS) = h_new; if dst_bf16 != null it gets bf16(h_new).
// ---------------------------------------------------------------------------
__device__ __forceinline__ void rk4_step(float* hbase, bf16_t* sA, bf16_t* sZ,
                                         float* ksum,
                                         const bf16_t* __restrict__ W1T,
                                         const bf16_t* sW2T,
                                         const float* __restrict__ b1,
                                         const float* __restrict__ b2,
                                         float t0, float t1, int wave, int lane,
                                         int tid, bf16_t* dst_bf16) {
  const float dt = t1 - t0;
  const float st[4] = {t0, t0 + 0.5f * dt, t0 + 0.5f * dt, t1};
  const float sc[4] = {0.5f * dt, 0.5f * dt, dt, 0.0f};
  const float sw[4] = {1.0f, 2.0f, 2.0f, 1.0f};
  const int z1 = opaque0();

  for (int i = tid; i < 16 * HID; i += 256) ksum[i] = 0.0f;
  __syncthreads();

#pragma unroll
  for (int s = 0; s < 4; ++s) {
    if (tid < 16) sA[tid * 160 + HID] = (bf16_t)st[s];  // time column
    __syncthreads();

    // z = tanh(sA @ W1T + b1)   (K = 160, 5 k-tiles)
    v8f zacc = {};
#pragma unroll
    for (int kt = 0; kt < 5; ++kt) {
      v16bf a = load_a_frag(sA, 160, kt * 32, lane);
      v16bf b = load_b_frag(W1T, z1, 5, kt, wave, lane);
      zacc = __builtin_amdgcn_wmma_f32_16x16x32_bf16(false, a, false, b,
                                                     (short)0, zacc, false, false);
    }
    {
      const int col = wave * 16 + (lane & 15);
      const int rb  = (lane >> 4) << 3;
      const float bb = b1[col];
#pragma unroll
      for (int r = 0; r < 8; ++r)
        sZ[(rb + r) * HID + col] = (bf16_t)tanhf(zacc[r] + bb);
    }
    __syncthreads();

    // k = sZ @ W2T + b2   (K = 128, 4 k-tiles, weights in LDS)
    v8f kacc = {};
#pragma unroll
    for (int kt = 0; kt < 4; ++kt) {
      v16bf a = load_a_frag(sZ, HID, kt * 32, lane);
      v16bf b = load_b_frag(sW2T, 0, 4, kt, wave, lane);
      kacc = __builtin_amdgcn_wmma_f32_16x16x32_bf16(false, a, false, b,
                                                     (short)0, kacc, false, false);
    }
    {
      const int col = wave * 16 + (lane & 15);
      const int rb  = (lane >> 4) << 3;
      const float bb = b2[col];
#pragma unroll
      for (int r = 0; r < 8; ++r) {
        const int row = rb + r;
        const float kv = kacc[r] + bb;
        ksum[row * HID + col] += sw[s] * kv;                 // unique (row,col) owner
        if (s < 3)
          sA[row * 160 + col] =
              (bf16_t)(hbase[row * HID + col] + sc[s] * kv); // next stage input
      }
    }
    __syncthreads();
  }

  const float w6 = dt * (1.0f / 6.0f);
  for (int i = tid; i < 16 * HID; i += 256) {
    const float hn = hbase[i] + w6 * ksum[i];
    hbase[i] = hn;
    if (dst_bf16) dst_bf16[i] = (bf16_t)hn;
  }
}

// ---------------------------------------------------------------------------
// Persistent main kernel: one workgroup per 16-row batch tile, full S scan.
// ---------------------------------------------------------------------------
__global__ void __launch_bounds__(256, 1)
node_lstm_main(const float* __restrict__ x, const float* __restrict__ tvec,
               const float* __restrict__ bih0, const float* __restrict__ bhh0,
               const float* __restrict__ nb1_0, const float* __restrict__ nb2_0,
               const float* __restrict__ bih1, const float* __restrict__ bhh1,
               const float* __restrict__ nb1_1, const float* __restrict__ nb2_1,
               const bf16_t* __restrict__ WihT0, const bf16_t* __restrict__ WhhT0,
               const bf16_t* __restrict__ WihT1, const bf16_t* __restrict__ WhhT1,
               const bf16_t* __restrict__ nW1T0, const bf16_t* __restrict__ nW2T0,
               const bf16_t* __restrict__ nW1T1, const bf16_t* __restrict__ nW2T1,
               float* __restrict__ y, float* __restrict__ hN,
               float* __restrict__ cN) {
  __shared__ float  G[16 * GDIM];                 // 32 KB gate scratch
  __shared__ float  hS0[16 * HID], cS0[16 * HID]; // layer-0 state
  __shared__ float  hS1[16 * HID], cS1[16 * HID]; // layer-1 state
  __shared__ float  ksum[16 * HID];               // RK4 accumulator
  __shared__ bf16_t sX[16 * INF];                 // layer-0 x staging
  __shared__ bf16_t sIn1[16 * HID];               // layer-1 input (= h0_new)
  __shared__ bf16_t sH[16 * HID];                 // current-layer h (bf16)
  __shared__ bf16_t sA[16 * 160];                 // fnode [h,t,0] staging
  __shared__ bf16_t sZ[16 * HID];                 // fnode hidden z
  __shared__ bf16_t sW2[2][8 * 4 * 512];          // RK4 W2^T frags, 2x32KB

  const int tid  = threadIdx.x;
  const int lane = tid & 31;
  const int wave = tid >> 5;
  const int b0   = blockIdx.x * 16;

  for (int i = tid; i < 16 * HID; i += 256) {
    hS0[i] = 0.0f; cS0[i] = 0.0f; hS1[i] = 0.0f; cS1[i] = 0.0f;
  }
  for (int i = tid; i < 16 * 160; i += 256) sA[i] = (bf16_t)0.0f;
  // Stage RK4 W2^T fragment arrays into LDS once (latency-critical phase).
  for (int i = tid; i < (8 * 4 * 512) / 8; i += 256) {
    ((v8bf*)sW2[0])[i] = ((const v8bf*)nW2T0)[i];
    ((v8bf*)sW2[1])[i] = ((const v8bf*)nW2T1)[i];
  }
  __syncthreads();

  float tprev = 0.0f;
#pragma unroll 1
  for (int t = 0; t < SEQ; ++t) {
    const float tcur = tvec[t];

    // Stage x_t tile (vectorized f32->bf16) and h0 (bf16)
    {
      const int m  = tid >> 4;
      const int k4 = (tid & 15) << 2;
      const float4 xv =
          *(const float4*)&x[((size_t)(b0 + m) * SEQ + t) * INF + k4];
      v4bf o;
      o[0] = (bf16_t)xv.x; o[1] = (bf16_t)xv.y;
      o[2] = (bf16_t)xv.z; o[3] = (bf16_t)xv.w;
      *(v4bf*)&sX[m * INF + k4] = o;
    }
    for (int i = tid; i < 16 * HID; i += 256) sH[i] = (bf16_t)hS0[i];
    if (t + 1 < SEQ)  // gfx1250 global_prefetch path for next timestep
      __builtin_prefetch(&x[((size_t)(b0 + (tid & 15)) * SEQ + (t + 1)) * INF],
                         0, 1);
    __syncthreads();

    // ---- layer 0 ----
    lstm_gates(sX, INF, 2, WihT0, sH, WhhT0, bih0, bhh0, G, wave, lane);
    __syncthreads();
    lstm_elem(G, hS0, cS0, sA, tid);
    __syncthreads();
    rk4_step(hS0, sA, sZ, ksum, nW1T0, sW2[0], nb1_0, nb2_0, tprev, tcur,
             wave, lane, tid, sIn1);
    __syncthreads();

    // ---- layer 1 ----
    for (int i = tid; i < 16 * HID; i += 256) sH[i] = (bf16_t)hS1[i];
    __syncthreads();
    lstm_gates(sIn1, HID, 4, WihT1, sH, WhhT1, bih1, bhh1, G, wave, lane);
    __syncthreads();
    lstm_elem(G, hS1, cS1, sA, tid);
    __syncthreads();
    rk4_step(hS1, sA, sZ, ksum, nW1T1, sW2[1], nb1_1, nb2_1, tprev, tcur,
             wave, lane, tid, nullptr);
    __syncthreads();

    // y[b, t, :] = h1
    for (int i = tid; i < 16 * HID; i += 256) {
      const int m = i >> 7, j = i & 127;
      y[((size_t)(b0 + m) * SEQ + t) * HID + j] = hS1[i];
    }
    tprev = tcur;
    __syncthreads();
  }

  // Final (hN, cN), layer-major (2, B, H)
  for (int i = tid; i < 16 * HID; i += 256) {
    const int m = i >> 7, j = i & 127;
    const size_t o = (size_t)(b0 + m) * HID + j;
    hN[o] = hS0[i];
    hN[(size_t)BATCH * HID + o] = hS1[i];
    cN[o] = cS0[i];
    cN[(size_t)BATCH * HID + o] = cS1[i];
  }
}

// ---------------------------------------------------------------------------
// Host launcher
// ---------------------------------------------------------------------------
extern "C" void kernel_launch(void* const* d_in, const int* in_sizes, int n_in,
                              void* d_out, int out_size, void* d_ws,
                              size_t ws_size, hipStream_t stream) {
  (void)in_sizes; (void)n_in; (void)out_size; (void)ws_size;
  const float* x     = (const float*)d_in[0];
  const float* tvec  = (const float*)d_in[1];
  const float* Wih0  = (const float*)d_in[2];
  const float* Whh0  = (const float*)d_in[3];
  const float* bih0  = (const float*)d_in[4];
  const float* bhh0  = (const float*)d_in[5];
  const float* nW1_0 = (const float*)d_in[6];
  const float* nb1_0 = (const float*)d_in[7];
  const float* nW2_0 = (const float*)d_in[8];
  const float* nb2_0 = (const float*)d_in[9];
  const float* Wih1  = (const float*)d_in[10];
  const float* Whh1  = (const float*)d_in[11];
  const float* bih1  = (const float*)d_in[12];
  const float* bhh1  = (const float*)d_in[13];
  const float* nW1_1 = (const float*)d_in[14];
  const float* nb1_1 = (const float*)d_in[15];
  const float* nW2_1 = (const float*)d_in[16];
  const float* nb2_1 = (const float*)d_in[17];

  bf16_t* ws = (bf16_t*)d_ws;
  size_t off = 0;
  bf16_t* WihT0 = ws + off; off += (size_t)(GDIM / 16) * (INF / 32) * 512;  // 32768
  bf16_t* WhhT0 = ws + off; off += (size_t)(GDIM / 16) * (HID / 32) * 512;  // 65536
  bf16_t* WihT1 = ws + off; off += (size_t)(GDIM / 16) * (HID / 32) * 512;
  bf16_t* WhhT1 = ws + off; off += (size_t)(GDIM / 16) * (HID / 32) * 512;
  bf16_t* W1T0  = ws + off; off += (size_t)(HID / 16) * (160 / 32) * 512;   // 20480
  bf16_t* W2T0  = ws + off; off += (size_t)(HID / 16) * (HID / 32) * 512;   // 16384
  bf16_t* W1T1  = ws + off; off += (size_t)(HID / 16) * (160 / 32) * 512;
  bf16_t* W2T1  = ws + off; off += (size_t)(HID / 16) * (HID / 32) * 512;

  struct PackJob { const float* W; bf16_t* dst; int Nout, Kin, Kpad; };
  const PackJob jobs[8] = {
      {Wih0,  WihT0, GDIM, INF, INF}, {Whh0,  WhhT0, GDIM, HID, HID},
      {Wih1,  WihT1, GDIM, HID, HID}, {Whh1,  WhhT1, GDIM, HID, HID},
      {nW1_0, W1T0,  HID, HID + 1, 160}, {nW2_0, W2T0, HID, HID, HID},
      {nW1_1, W1T1,  HID, HID + 1, 160}, {nW2_1, W2T1, HID, HID, HID},
  };
  for (int j = 0; j < 8; ++j) {
    const int total = (jobs[j].Nout / 16) * (jobs[j].Kpad / 32) * 512;
    pack_wT<<<dim3((total + 255) / 256), dim3(256), 0, stream>>>(
        jobs[j].W, jobs[j].dst, jobs[j].Nout, jobs[j].Kin, jobs[j].Kpad);
  }

  float* y  = (float*)d_out;
  float* hN = y + (size_t)BATCH * SEQ * HID;
  float* cN = hN + (size_t)2 * BATCH * HID;

  node_lstm_main<<<dim3(BATCH / 16), dim3(256), 0, stream>>>(
      x, tvec, bih0, bhh0, nb1_0, nb2_0, bih1, bhh1, nb1_1, nb2_1,
      WihT0, WhhT0, WihT1, WhhT1, W1T0, W2T0, W1T1, W2T1, y, hN, cN);
}
